// Corr_pyTorch_80917183856845
// MI455X (gfx1250) — matrix operand order repaired and checked by reading, not compiled, not run on hardware
//
#include <hip/hip_runtime.h>

typedef __attribute__((ext_vector_type(16))) _Float16 v16h;
typedef __attribute__((ext_vector_type(8)))  _Float16 v8h;
typedef __attribute__((ext_vector_type(2)))  _Float16 v2h;
typedef __attribute__((ext_vector_type(8)))  float    v8f;
typedef __attribute__((ext_vector_type(4)))  float    v4f;

namespace {
constexpr int kB = 8, kC = 96, kH = 160, kW = 320;
constexpr int kHW = kH * kW;
constexpr int kPad = 4;           // max displacement
constexpr int kNd = 9;            // displacements per axis
constexpr int kNch = 81;          // output channels
constexpr int kTY = 8;            // y rows per workgroup (one per wave)
constexpr int kTX = 16;           // x cols per workgroup (WMMA M)
constexpr int kYS = kTY + 2 * kPad;   // 16 staged in2 rows
constexpr int kXS = 32;               // staged in2 cols: x0-4 .. x0+27
constexpr int kAElems = kTY * kTX * kC;   // 12288 f16 = 24 KB
constexpr int kBElems = kYS * kXS * kC;   // 49152 f16 = 96 KB
constexpr int kDStride = 20;              // padded D row (floats): no LDS bank conflicts
constexpr int kDWave   = 32 * kDStride;   // 640 floats per wave region (fits in sA space)
}

// Build a 16-half WMMA fragment from two contiguous 16B LDS reads (ds_load_b128).
__device__ __forceinline__ v16h load_frag(const _Float16* p0, const _Float16* p1) {
  v8h lo = *(const v8h*)p0;
  v8h hh = *(const v8h*)p1;
  return __builtin_shufflevector(lo, hh, 0, 1, 2, 3, 4, 5, 6, 7,
                                         8, 9, 10, 11, 12, 13, 14, 15);
}

__global__ __launch_bounds__(256)
void corr81_wmma(const float* __restrict__ in1, const float* __restrict__ in2,
                 float* __restrict__ out) {
  __shared__ __align__(32) unsigned char smem[(kAElems + kBElems) * 2];
  _Float16* sA = (_Float16*)smem;                   // [y][x][c], retired after frag load
  _Float16* sB = (_Float16*)(smem + kAElems * 2);   // [yy][xn][c]
  float*    sD = (float*)smem;                      // reused: per-wave D tiles [n][m]

  const int x0 = blockIdx.x * kTX;
  const int y0 = blockIdx.y * kTY;
  const int b  = blockIdx.z;
  const int t  = threadIdx.x;

  // ---- stage in1 tile: f32 global (coalesced over x, nontemporal) -> f16 LDS ----
  {
    const int tx = t & (kTX - 1);
    const int tg = t >> 4;                 // 16 groups
    const int y  = tg & 7;
    const int c0 = (tg >> 3) * (kC / 2);   // 0 or 48
    const float* src = in1 + (((size_t)b * kC + c0) * kH + (y0 + y)) * kW + (x0 + tx);
    _Float16* dst = sA + (y * kTX + tx) * kC + c0;
#pragma unroll 4
    for (int c = 0; c < kC / 2; c += 2) {
      float u0 = __builtin_nontemporal_load(src + (size_t)c * kHW);
      float u1 = __builtin_nontemporal_load(src + (size_t)(c + 1) * kHW);
      v2h pk; pk[0] = (_Float16)u0; pk[1] = (_Float16)u1;
      *(v2h*)(dst + c) = pk;
    }
  }
  // ---- stage in2 halo tile: zero-padded -> f16 LDS [yy][xn][c] ----
  {
    const int tx = t & (kXS - 1);          // 0..31, coalesced over x
    const int tg = t >> 5;                 // 0..7
    const int gx = x0 - kPad + tx;
    const bool xok = (gx >= 0) && (gx < kW);
#pragma unroll
    for (int yi = 0; yi < 2; ++yi) {
      const int yy = tg * 2 + yi;          // 0..15
      const int gy = y0 - kPad + yy;
      _Float16* dst = sB + (yy * kXS + tx) * kC;
      if (xok && gy >= 0 && gy < kH) {
        const float* src = in2 + (((size_t)b * kC) * kH + gy) * kW + gx;
#pragma unroll 4
        for (int c = 0; c < kC; c += 2) {
          float u0 = src[(size_t)c * kHW];
          float u1 = src[(size_t)(c + 1) * kHW];
          v2h pk; pk[0] = (_Float16)u0; pk[1] = (_Float16)u1;
          *(v2h*)(dst + c) = pk;
        }
      } else {
        const v2h z = {};
#pragma unroll 8
        for (int c = 0; c < kC; c += 2) *(v2h*)(dst + c) = z;
      }
    }
  }
  __syncthreads();

  // ---- per-wave GEMM: wave w owns output row y0+w ----
  const int wave = t >> 5;
  const int lane = t & 31;
  const int lm   = lane & 15;
  const int hi   = lane >> 4;

  // A fragments (3 K-chunks of 32) resident across the whole dy loop.
  // A 16x32: all lanes M=lane%16; elements K=16*hi..16*hi+15 (contiguous, matches B).
  const _Float16* aRow = sA + (wave * kTX + lm) * kC;
  v16h afrag[3];
#pragma unroll
  for (int kc = 0; kc < 3; ++kc) {
    const int k0 = kc * 32;
    afrag[kc] = load_frag(aRow + k0 + hi * 16, aRow + k0 + hi * 16 + 8);
  }
  __syncthreads();   // all waves done reading sA -> safe to reuse as sD

  float* dW = sD + wave * kDWave;          // wave-private D staging region
  const float scale = 1.0f / (float)kC;
  const int obase = b * (kNch * kHW) + (y0 + wave) * kW + x0;

#pragma unroll
  for (int dy = 0; dy < kNd; ++dy) {       // dy index 0..8 <-> -4..+4
    const int yy = wave + dy;              // staged in2 row
    // B 32x16 in this code's order: lane column N=lane%16, K = 16*hi + 2v+p
    const _Float16* bRow = sB + (yy * kXS) * kC + hi * 16;
    v8f acc0 = {};  // source cols x0-4 .. x0+11
    v8f acc1 = {};  // source cols x0+12 .. x0+27
#pragma unroll
    for (int kc = 0; kc < 3; ++kc) {
      const int k0 = kc * 32;
      const v16h b0 = load_frag(bRow + lm * kC + k0, bRow + lm * kC + k0 + 8);
      const v16h b1 = load_frag(bRow + (lm + 16) * kC + k0,
                                bRow + (lm + 16) * kC + k0 + 8);
      acc0 = __builtin_amdgcn_wmma_f32_16x16x32_f16(
          false, afrag[kc], false, b0, (short)0, acc0, false, false);
      acc1 = __builtin_amdgcn_wmma_f32_16x16x32_f16(
          false, afrag[kc], false, b1, (short)0, acc1, false, false);
    }

    // Stage D tiles to LDS as [n][m] (row stride 20 floats, conflict-free b128).
    // Lane holds column n = lm (+16), rows m = 8*hi + 0..7 -> two contiguous float4.
    {
      v4f a0lo = __builtin_shufflevector(acc0, acc0, 0, 1, 2, 3);
      v4f a0hi = __builtin_shufflevector(acc0, acc0, 4, 5, 6, 7);
      v4f a1lo = __builtin_shufflevector(acc1, acc1, 0, 1, 2, 3);
      v4f a1hi = __builtin_shufflevector(acc1, acc1, 4, 5, 6, 7);
      float* p0 = dW + lm * kDStride + hi * 8;
      float* p1 = dW + (lm + 16) * kDStride + hi * 8;
      *(v4f*)(p0)     = a0lo;
      *(v4f*)(p0 + 4) = a0hi;
      *(v4f*)(p1)     = a1lo;
      *(v4f*)(p1 + 4) = a1hi;
    }

    // Coalesced write-out: 144 = 9 dx * 16 x values along the tile diagonal.
    // Half-wave writes 16 contiguous dwords of one (ch, y) row. Nontemporal:
    // output is write-once, keep it out of L2 (in2 is re-read).
#pragma unroll
    for (int j = 0; j < 5; ++j) {
      const int i = lane + j * 32;
      if (i < kNd * kTX) {
        const int dxi = i >> 4;            // 0..8
        const int m   = i & 15;
        const float val = dW[(m + dxi) * kDStride + m] * scale;
        __builtin_nontemporal_store(val, out + obase + (dy * kNd + dxi) * kHW + m);
      }
    }
  }
}

extern "C" void kernel_launch(void* const* d_in, const int* in_sizes, int n_in,
                              void* d_out, int out_size, void* d_ws, size_t ws_size,
                              hipStream_t stream) {
  const float* in1 = (const float*)d_in[0];
  const float* in2 = (const float*)d_in[1];
  float* out = (float*)d_out;
  dim3 grid(kW / kTX, kH / kTY, kB);   // (20, 20, 8)
  dim3 block(256);                     // 8 waves (wave32)
  hipLaunchKernelGGL(corr81_wmma, grid, block, 0, stream, in1, in2, out);
}
